// ContextualAttentionModule_29884382446348
// MI455X (gfx1250) — matrix operand order repaired and checked by reading, compiled, not run on hardware
//
#include <hip/hip_runtime.h>

typedef __attribute__((ext_vector_type(16))) _Float16 v16h;
typedef __attribute__((ext_vector_type(8)))  _Float16 v8h;
typedef __attribute__((ext_vector_type(8)))  float    v8f;

#define Hh 64
#define Ww 64
#define Lp 4096      // H*W
#define Cc 64
#define CK 576       // C*9
#define EPSF 1e-7f

// ---------------------------------------------------------------------------
// prep_k: build normalized patch kernels Kf16[l][576] and transpose Kt[576][l]
// one block (64 threads) per location l; thread t owns channel c=t (9 taps)
// ---------------------------------------------------------------------------
__global__ __launch_bounds__(64) void prep_k_kernel(
    const float* __restrict__ fg, const float* __restrict__ mask,
    _Float16* __restrict__ Kf, _Float16* __restrict__ Kt)
{
  int l = blockIdx.x;
  int y = l >> 6, x = l & 63;
  int t = threadIdx.x;           // channel
  float v[9];
  float ss = 0.f;
#pragma unroll
  for (int i = 0; i < 3; ++i)
#pragma unroll
    for (int j = 0; j < 3; ++j) {
      int yy = y + i - 1, xx = x + j - 1;
      float f = 0.f;
      if (yy >= 0 && yy < Hh && xx >= 0 && xx < Ww)
        f = fg[t * Lp + yy * Ww + xx] * mask[yy * Ww + xx];
      f += EPSF;                 // reference adds EPS to padded patch too
      v[i * 3 + j] = f;
      ss += f * f;
    }
  __shared__ float red[64];
  red[t] = ss;
  __syncthreads();
#pragma unroll
  for (int s = 32; s > 0; s >>= 1) {
    if (t < s) red[t] += red[t + s];
    __syncthreads();
  }
  float rn = 1.0f / sqrtf(red[0]);
#pragma unroll
  for (int q = 0; q < 9; ++q) {
    _Float16 h = (_Float16)(v[q] * rn);
    int ck = t * 9 + q;
    Kf[l * CK + ck] = h;         // A operand of GEMM1 (k-contiguous)
    Kt[ck * Lp + l] = h;         // B^T operand of GEMM2 (k-contiguous)
  }
}

// ---------------------------------------------------------------------------
// prep_f: unfold foreground into FT[p][576] f16 (B^T operand of GEMM1)
// ---------------------------------------------------------------------------
__global__ __launch_bounds__(256) void prep_f_kernel(
    const float* __restrict__ fg, _Float16* __restrict__ FT)
{
  int idx = blockIdx.x * 256 + threadIdx.x;   // < Lp*Cc
  int c = idx & 63;
  int p = idx >> 6;
  int y = p >> 6, x = p & 63;
#pragma unroll
  for (int i = 0; i < 3; ++i)
#pragma unroll
    for (int j = 0; j < 3; ++j) {
      int yy = y + i - 1, xx = x + j - 1;
      float f = 0.f;
      if (yy >= 0 && yy < Hh && xx >= 0 && xx < Ww)
        f = fg[c * Lp + yy * Ww + xx];
      FT[p * CK + c * 9 + i * 3 + j] = (_Float16)f;
    }
}

// ---------------------------------------------------------------------------
// WMMA GEMM:  C[M][N] = A[M][K] * B^T[N][K]   (f16 in, f32 accumulate)
// 256 threads = 8 waves stacked in M; wave tile 64x64 = 4x4 WMMA tiles
// (8 fragment loads -> 16 WMMAs per k-step: 1 b128 load per WMMA).
// Block tile 512(M) x 64(N). All dims divisible -> no edge handling.
// Operands are L2-resident on MI455X (192MB) -> direct global fragment loads.
// ---------------------------------------------------------------------------
__device__ __forceinline__ v16h cat8(v8h lo, v8h hi) {
  return __builtin_shufflevector(lo, hi, 0, 1, 2, 3, 4, 5, 6, 7,
                                 8, 9, 10, 11, 12, 13, 14, 15);
}

__global__ __launch_bounds__(256) void wmma_gemm_nt(
    const _Float16* __restrict__ A, int lda,
    const _Float16* __restrict__ BT, int ldb,
    float* __restrict__ Cm, int ldc, int Kdim)
{
  int lane = threadIdx.x & 31;
  int wave = threadIdx.x >> 5;       // 0..7 stacked in M
  int m0 = blockIdx.y * 512 + wave * 64;
  int n0 = blockIdx.x * 64;

  // Fragment addressing chosen here (wave32; A and B share this k order):
  // A 16x32 f16: lane L -> row M = L&15 ; 16 contiguous k at k0+(L>=16?16:0).
  // B 32x16 f16: lane L -> col N = L&15 ; 16 contiguous k at k0+(L>=16?16:0).
  int arow = m0 + (lane & 15);
  int akoff = (lane >> 4) << 4;      // 0 or 16
  int brow = n0 + (lane & 15);
  int bkoff = (lane >> 4) << 4;      // 0 or 16

  v8f acc[4][4];
#pragma unroll
  for (int i = 0; i < 4; ++i)
#pragma unroll
    for (int j = 0; j < 4; ++j) acc[i][j] = (v8f){};

  for (int k = 0; k < Kdim; k += 32) {
    v16h a[4], b[4];
#pragma unroll
    for (int i = 0; i < 4; ++i) {
      const _Float16* ap = A + (long)(arow + 16 * i) * lda + k + akoff;
      a[i] = cat8(*(const v8h*)ap, *(const v8h*)(ap + 8));
    }
#pragma unroll
    for (int j = 0; j < 4; ++j) {
      const _Float16* bp = BT + (long)(brow + 16 * j) * ldb + k + bkoff;
      b[j] = *(const v16h*)bp;
    }
#pragma unroll
    for (int i = 0; i < 4; ++i)
#pragma unroll
      for (int j = 0; j < 4; ++j)
        acc[i][j] = __builtin_amdgcn_wmma_f32_16x16x32_f16(
            false, a[i], false, b[j], (short)0, acc[i][j], false, false);
  }

  // D layout: lane L -> col N = L&15 ; VGPR r -> row M = r + (L>=16 ? 8 : 0)
  int crow = (lane >> 4) << 3;
  int ccol = lane & 15;
#pragma unroll
  for (int i = 0; i < 4; ++i)
#pragma unroll
    for (int j = 0; j < 4; ++j) {
      float* cp = Cm + (long)(m0 + 16 * i + crow) * ldc + (n0 + 16 * j + ccol);
#pragma unroll
      for (int r = 0; r < 8; ++r) cp[(long)r * ldc] = acc[i][j][r];
    }
}

// ---------------------------------------------------------------------------
// boxsum: 3x3 SAME box filter over each 64x64 score map (one block per l)
// ---------------------------------------------------------------------------
__global__ __launch_bounds__(256) void boxsum_kernel(
    const float* __restrict__ S, float* __restrict__ T)
{
  int l = blockIdx.x;
  __shared__ float tile[66][66];
  int t = threadIdx.x;
  for (int i = t; i < 66 * 66; i += 256) ((float*)tile)[i] = 0.f;
  __syncthreads();
  for (int i = t; i < 4096; i += 256)
    tile[(i >> 6) + 1][(i & 63) + 1] = S[(long)l * Lp + i];
  __syncthreads();
  for (int i = t; i < 4096; i += 256) {
    int y = i >> 6, x = i & 63;
    float s = 0.f;
#pragma unroll
    for (int di = 0; di < 3; ++di)
#pragma unroll
      for (int dj = 0; dj < 3; ++dj)
        s += tile[y + di][x + dj];
    T[(long)l * Lp + i] = s;
  }
}

// ---------------------------------------------------------------------------
// softmax over l for each pixel p; writes attn^T[p][l] as f16 via an LDS
// transpose tile so GEMM2's A operand is k-contiguous.
// grid = 64 blocks, block owns pixel columns p0..p0+63; 256 thr = 64px x 4 slices
// ---------------------------------------------------------------------------
__global__ __launch_bounds__(256) void softmax_t_kernel(
    float* __restrict__ T, _Float16* __restrict__ attnT)
{
  int t = threadIdx.x;
  int px = t & 63;
  int sl = t >> 6;                  // 0..3
  int p0 = blockIdx.x * 64;
  __shared__ float red[4 * 64];
  __shared__ float colv[64];
  __shared__ _Float16 th[64][66];

  // pass 1: column max over l
  float m = -3.4e38f;
  for (int i = sl; i < Lp; i += 4) m = fmaxf(m, T[(long)i * Lp + p0 + px]);
  red[sl * 64 + px] = m;
  __syncthreads();
  if (sl == 0)
    colv[px] = fmaxf(fmaxf(red[px], red[64 + px]),
                     fmaxf(red[128 + px], red[192 + px]));
  __syncthreads();
  float cm = colv[px];
  __syncthreads();

  // pass 2: exp in place + column sum
  float s = 0.f;
  for (int i = sl; i < Lp; i += 4) {
    long idx = (long)i * Lp + p0 + px;
    float v = __expf(T[idx] - cm);
    T[idx] = v;
    s += v;
  }
  red[sl * 64 + px] = s;
  __syncthreads();
  if (sl == 0)
    colv[px] = 1.f / (red[px] + red[64 + px] + red[128 + px] + red[192 + px]);
  __syncthreads();
  float ri = colv[px];

  // pass 3: normalize + transpose 64x64 chunks -> attnT[p][l] f16
  for (int lc = 0; lc < 64; ++lc) {
    int l0 = lc * 64;
#pragma unroll
    for (int q = 0; q < 16; ++q) {
      int li = sl * 16 + q;
      th[px][li] = (_Float16)(T[(long)(l0 + li) * Lp + p0 + px] * ri);
    }
    __syncthreads();
    int row = t >> 2, seg = t & 3;
#pragma unroll
    for (int q = 0; q < 16; ++q)
      attnT[(long)(p0 + row) * Lp + l0 + seg * 16 + q] = th[row][seg * 16 + q];
    __syncthreads();
  }
}

// ---------------------------------------------------------------------------
// scatter: transposed-conv accumulation of PoutT[p][ck] + mask blend
// ---------------------------------------------------------------------------
__global__ __launch_bounds__(256) void scatter_kernel(
    const float* __restrict__ fg, const float* __restrict__ mask,
    const float* __restrict__ PoutT, float* __restrict__ out)
{
  int idx = blockIdx.x * 256 + threadIdx.x;   // < Cc*Lp
  int x = idx & 63;
  int y = (idx >> 6) & 63;
  int c = idx >> 12;
  float mk = mask[y * Ww + x];
  float s = 0.f;
#pragma unroll
  for (int dy = 0; dy < 3; ++dy)
#pragma unroll
    for (int dx = 0; dx < 3; ++dx) {
      int yy = y - dy + 1, xx = x - dx + 1;
      if (yy >= 0 && yy < Hh && xx >= 0 && xx < Ww)
        s += PoutT[(long)(yy * Ww + xx) * CK + c * 9 + dy * 3 + dx];
    }
  out[c * Lp + y * Ww + x] =
      fg[c * Lp + y * Ww + x] * mk + (1.f - mk) * s * (1.f / 9.f);
}

// ---------------------------------------------------------------------------
extern "C" void kernel_launch(void* const* d_in, const int* in_sizes, int n_in,
                              void* d_out, int out_size, void* d_ws, size_t ws_size,
                              hipStream_t stream) {
  const float* fg   = (const float*)d_in[0];   // [4,64,64,64]
  const float* mask = (const float*)d_in[1];   // [4,1,64,64]
  float* out = (float*)d_out;

  char* w = (char*)d_ws;
  _Float16* Kf = (_Float16*)w;            w += (size_t)Lp * CK * 2;   // 4.7 MB
  _Float16* Kt = (_Float16*)w;            w += (size_t)CK * Lp * 2;   // 4.7 MB
  _Float16* FT = (_Float16*)w;            w += (size_t)Lp * CK * 2;   // 4.7 MB
  float*    S  = (float*)w;               w += (size_t)Lp * Lp * 4;   // 64 MB
  float*    T  = (float*)w;               w += (size_t)Lp * Lp * 4;   // 64 MB
  float*    PT = (float*)w;               w += (size_t)Lp * CK * 4;   // 9.4 MB
  _Float16* attnT = (_Float16*)S;         // S is dead after boxsum -> alias

  for (int b = 0; b < 4; ++b) {
    const float* fgb = fg + (size_t)b * Cc * Lp;
    const float* mb  = mask + (size_t)b * Lp;
    float* ob = out + (size_t)b * Cc * Lp;

    prep_k_kernel<<<Lp, 64, 0, stream>>>(fgb, mb, Kf, Kt);
    prep_f_kernel<<<(Lp * Cc) / 256, 256, 0, stream>>>(fgb, FT);
    // GEMM1: S[l][p] = Kf[l][ck] . FT[p][ck]   (M=4096,N=4096,K=576)
    wmma_gemm_nt<<<dim3(Lp / 64, Lp / 512), 256, 0, stream>>>(
        Kf, CK, FT, CK, S, Lp, CK);
    boxsum_kernel<<<Lp, 256, 0, stream>>>(S, T);
    softmax_t_kernel<<<64, 256, 0, stream>>>(T, attnT);
    // GEMM2: PT[p][ck] = attnT[p][l] . Kt[ck][l]  (M=4096,N=576,K=4096)
    wmma_gemm_nt<<<dim3(CK / 64, Lp / 512), 256, 0, stream>>>(
        attnT, Lp, Kt, Lp, PT, CK, Lp);
    scatter_kernel<<<(Cc * Lp) / 256, 256, 0, stream>>>(fgb, mb, PT, ob);
  }
}
